// SlidingTileAttention_52974126629419
// MI455X (gfx1250) — matrix operand order repaired and checked by reading, not compiled, not run on hardware
//
#include <hip/hip_runtime.h>
#include <hip/hip_bf16.h>
#include <cstdint>

// ---------------- CDNA5 WMMA plumbing (wave32, 16x16x32 bf16, f32 accum) ---
typedef __bf16 v16bf __attribute__((ext_vector_type(16)));
typedef __bf16 v8bf  __attribute__((ext_vector_type(8)));
typedef float  v8f   __attribute__((ext_vector_type(8)));

#define DEVI __device__ __forceinline__

DEVI __bf16 f2bf(float f) {
    union { float f; uint32_t u; } a; a.f = f;
    uint32_t r = a.u + 0x7FFFu + ((a.u >> 16) & 1u);   // round-to-nearest-even
    union { uint16_t s; __bf16 b; } o; o.s = (uint16_t)(r >> 16);
    return o.b;
}

DEVI v8f vzero() { v8f c;
#pragma unroll
    for (int i = 0; i < 8; ++i) c[i] = 0.f;
    return c; }

// A fragment: 16x32 (MxK), bf16, row-major source with leading dim ld.
// Per ISA 7.12.2: lane L (hv=L>>4, row=L&15): VGPR0-3 hold K = hv*8 .. hv*8+7,
// VGPR4-7 hold K = 16+hv*8 .. 16+hv*8+7  -> two contiguous 8-elem chunks.
DEVI v16bf load_afrag(const __bf16* __restrict__ base, int ld, int row0, int k0, int lane) {
    const int hv = lane >> 4;
    const __bf16* p = base + (size_t)(row0 + (lane & 15)) * ld + k0 + hv * 8;
    v8bf lo = *(const v8bf*)(p);
    v8bf hi = *(const v8bf*)(p + 16);
    v16bf r;
#pragma unroll
    for (int e = 0; e < 8; ++e) { r[e] = lo[e]; r[e + 8] = hi[e]; }
    return r;
}

// B fragment: 32x16 (KxN) where B[k][n] = W[n][k] (i.e. we compute act @ W^T).
// Lane L: col n = n0 + (L&15), K = k0 + (L>>4)*8 .. +7 and k0 + 16 + (L>>4)*8 .. +7
// straight from row n of the row-major bf16 weight matrix (no transpose).
DEVI v16bf load_bfrag(const __bf16* __restrict__ base, int ld, int n0, int k0, int lane) {
    const __bf16* p = base + (size_t)(n0 + (lane & 15)) * ld + k0 + (lane >> 4) * 8;
    v8bf lo = *(const v8bf*)(p);
    v8bf hi = *(const v8bf*)(p + 16);
    v16bf r;
#pragma unroll
    for (int e = 0; e < 8; ++e) { r[e] = lo[e]; r[e + 8] = hi[e]; }
    return r;
}

DEVI v8f wmma_bf16(v16bf a, v16bf b, v8f c) {
    return __builtin_amdgcn_wmma_f32_16x16x32_bf16(false, a, false, b, (short)0, c,
                                                   false, false);
}

// C/D 16x16 f32 tile: element e of v8f = (row0 + e + (lane>>4)*8, col0 + (lane&15))
template <int KDIM>
DEVI v8f gemm_tile(const __bf16* __restrict__ A, int lda, int row0,
                   const __bf16* __restrict__ B, int ldb, int n0, int lane) {
    v8f c = vzero();
    for (int k0 = 0; k0 < KDIM; k0 += 32) {
        if (k0 + 32 < KDIM) __builtin_prefetch(B + (size_t)(n0 + (lane & 15)) * ldb + k0 + 32, 0, 1);
        c = wmma_bf16(load_afrag(A, lda, row0, k0, lane),
                      load_bfrag(B, ldb, n0, k0, lane), c);
    }
    return c;
}

// ---- CDNA5 async global->LDS staging (ASYNCcnt path, no VGPR round-trip) ----
// Issues GLOBAL_LOAD_ASYNC_TO_LDS_B128 per 16B chunk, then fences with
// s_wait_asynccnt 0 + workgroup barrier. Flat pointer low 32 bits == LDS offset.
DEVI void stage_async(__bf16* dst_lds, const __bf16* src, int halves, int tid) {
    const int chunks = halves >> 3;                 // 8 bf16 = 16B per transfer
    for (int c = tid; c < chunks; c += 256) {
        uint32_t loff = (uint32_t)(uintptr_t)(dst_lds + c * 8);
        const __bf16* g = src + c * 8;
        asm volatile("global_load_async_to_lds_b128 %0, %1, off"
                     :: "v"(loff), "v"(g) : "memory");
    }
    asm volatile("s_wait_asynccnt 0x0" ::: "memory");
    __syncthreads();
}

// ---------------- Problem constants ----------------
// x: (1, T=16, H=64, W=64, D=512); tiles 8x8 -> S=64 tokens, NTILE=1024
// local: 8 heads x 64;  global: L=1024, 4 heads x 64, GDIM=256

// K0: fp32 -> bf16 conversion (weights)
__global__ void k_cvt(const float* __restrict__ src, __bf16* __restrict__ dst, int n) {
    int i = blockIdx.x * 256 + threadIdx.x;
    if (i < n) dst[i] = f2bf(src[i]);
}

// K1: gather x into per-tile rows (bf16) + fp32 tile means -> bf16
__global__ __launch_bounds__(256) void k_tiles(const float* __restrict__ x,
                                               __bf16* __restrict__ xt,
                                               __bf16* __restrict__ xf) {
    const int tile = blockIdx.x;                 // ((t*8+i)*8+j) == L index
    const int t = tile >> 6, i = (tile >> 3) & 7, j = tile & 7;
#pragma unroll
    for (int c = 0; c < 2; ++c) {
        const int d = c * 256 + threadIdx.x;     // coalesced over d
        float sum = 0.f;
        for (int s = 0; s < 64; ++s) {
            const int hh = i * 8 + (s >> 3), ww = j * 8 + (s & 7);
            float v = x[(((size_t)t * 64 + hh) * 64 + ww) * 512 + d];
            sum += v;
            xt[((size_t)tile * 64 + s) * 512 + d] = f2bf(v);
        }
        xf[(size_t)tile * 512 + d] = f2bf(sum * (1.f / 64.f));
    }
}

// K2: fused local attention per tile. The 64x512 x-tile is staged ONCE in LDS
// via async-LDS loads and feeds the A side of all 8 heads' QKV GEMMs (the tile
// is head-invariant) -- removes ~2 GB of redundant L2 fragment traffic.
// Static LDS = 112 KB; gfx1250 WGP has 320 KB -> 2 workgroups resident.
__global__ __launch_bounds__(256) void k_local_attn(
    const __bf16* __restrict__ xt, const __bf16* __restrict__ wl,
    const float* __restrict__ bl, __bf16* __restrict__ o_ws,
    const float* __restrict__ scale_p) {
    __shared__ alignas(32) __bf16 xs[64 * 512];     // staged x tile (64 KB)
    __shared__ alignas(32) __bf16 qs[64 * 64];
    __shared__ alignas(32) __bf16 ks[64 * 64];
    __shared__ alignas(32) __bf16 vts[64 * 64];     // v transposed: [d][t]
    __shared__ alignas(32) __bf16 Ps[64 * 64];
    __shared__ alignas(16) float  Ss[64 * 64];

    const int tile = blockIdx.x;
    const int tid = threadIdx.x, lane = tid & 31, wave = tid >> 5;
    const float scale = scale_p[0];

    stage_async(xs, xt + (size_t)tile * 64 * 512, 64 * 512, tid);

    for (int h = 0; h < 8; ++h) {
        // ---- QKV for head h: 3 matrices x (4x4) 16x16 tiles = 48, 6 per wave
        v8f acc[6];
#pragma unroll
        for (int u = 0; u < 6; ++u) acc[u] = vzero();
        for (int k0 = 0; k0 < 512; k0 += 32) {
#pragma unroll
            for (int u = 0; u < 6; ++u) {
                const int idx = wave + 8 * u, mat = idx >> 4, r = idx & 15;
                const int mt = r >> 2, nt = r & 3;
                v16bf a = load_afrag(xs, 512, mt * 16, k0, lane);
                v16bf b = load_bfrag(wl, 512, mat * 512 + h * 64 + nt * 16, k0, lane);
                acc[u] = wmma_bf16(a, b, acc[u]);
            }
        }
#pragma unroll
        for (int u = 0; u < 6; ++u) {
            const int idx = wave + 8 * u, mat = idx >> 4, r = idx & 15;
            const int mt = r >> 2, nt = r & 3;
            const int col = nt * 16 + (lane & 15);
            const float bias = bl[mat * 512 + h * 64 + col];
#pragma unroll
            for (int e = 0; e < 8; ++e) {
                const int row = mt * 16 + (lane >> 4) * 8 + e;
                __bf16 val = f2bf(acc[u][e] + bias);
                if (mat == 0)      qs[row * 64 + col] = val;
                else if (mat == 1) ks[row * 64 + col] = val;
                else               vts[col * 64 + row] = val;   // transpose v
            }
        }
        __syncthreads();

        // ---- S = q @ k^T * scale, clip(+-10000): 16 tiles, 2 per wave
#pragma unroll
        for (int u = 0; u < 2; ++u) {
            const int idx = wave + 8 * u, mt = idx >> 2, nt = idx & 3;
            v8f s = gemm_tile<64>(qs, 64, mt * 16, ks, 64, nt * 16, lane);
            const int col = nt * 16 + (lane & 15);
#pragma unroll
            for (int e = 0; e < 8; ++e) {
                const int row = mt * 16 + (lane >> 4) * 8 + e;
                float sv = s[e] * scale;
                sv = fminf(fmaxf(sv, -10000.f), 10000.f);
                Ss[row * 64 + col] = sv;
            }
        }
        __syncthreads();

        // ---- softmax: 4 lanes per row (lanes adjacent within a wave32)
        {
            const int row = tid >> 2, q = tid & 3;
            const float* sr = Ss + row * 64 + q * 16;
            float m = -1e30f;
#pragma unroll
            for (int e = 0; e < 16; ++e) m = fmaxf(m, sr[e]);
            m = fmaxf(m, __shfl_xor(m, 1, 32));
            m = fmaxf(m, __shfl_xor(m, 2, 32));
            float p[16], sum = 0.f;
#pragma unroll
            for (int e = 0; e < 16; ++e) { p[e] = __expf(sr[e] - m); sum += p[e]; }
            sum += __shfl_xor(sum, 1, 32);
            sum += __shfl_xor(sum, 2, 32);
            const float inv = 1.f / sum;       // diag term guarantees sum >= ~1
            __bf16* pr = Ps + row * 64 + q * 16;
#pragma unroll
            for (int e = 0; e < 16; ++e) pr[e] = f2bf(p[e] * inv);
        }
        __syncthreads();

        // ---- O_h = P @ V: 16 tiles, 2 per wave; write head block to o_ws
#pragma unroll
        for (int u = 0; u < 2; ++u) {
            const int idx = wave + 8 * u, mt = idx >> 2, nt = idx & 3;
            v8f o = gemm_tile<64>(Ps, 64, mt * 16, vts, 64, nt * 16, lane);
            const int col = nt * 16 + (lane & 15);
#pragma unroll
            for (int e = 0; e < 8; ++e) {
                const int row = mt * 16 + (lane >> 4) * 8 + e;
                o_ws[((size_t)tile * 64 + row) * 512 + h * 64 + col] = f2bf(o[e]);
            }
        }
        __syncthreads();   // before next head reuses LDS
    }
}

// K4: global QKV GEMM [1024,512]@[512,768]^T; A-slab staged in LDS (async)
__global__ __launch_bounds__(256) void k_gqkv(const __bf16* __restrict__ xf,
                                              const __bf16* __restrict__ wg,
                                              const float* __restrict__ bg,
                                              __bf16* __restrict__ qg,
                                              __bf16* __restrict__ kg,
                                              __bf16* __restrict__ vgT) {
    __shared__ alignas(32) __bf16 xs[64 * 512];
    const int mb = blockIdx.x, nb = blockIdx.y;
    const int tid = threadIdx.x, lane = tid & 31, wave = tid >> 5;
    stage_async(xs, xf + (size_t)mb * 64 * 512, 64 * 512, tid);
#pragma unroll
    for (int u = 0; u < 8; ++u) {
        const int idx = wave + 8 * u, mt = idx >> 4, nt = idx & 15;
        const int n0 = nb * 256 + nt * 16;
        v8f c = gemm_tile<512>(xs, 512, mt * 16, wg, 512, n0, lane);
        const int n = n0 + (lane & 15);
        const float bias = bg[n];
        const int head = (n >> 6) & 3, d = n & 63, sect = n >> 8;
#pragma unroll
        for (int e = 0; e < 8; ++e) {
            const int m = mb * 64 + mt * 16 + (lane >> 4) * 8 + e;
            __bf16 val = f2bf(c[e] + bias);
            if (sect == 0)      qg[((size_t)head * 1024 + m) * 64 + d] = val;
            else if (sect == 1) kg[((size_t)head * 1024 + m) * 64 + d] = val;
            else                vgT[((size_t)head * 64 + d) * 1024 + m] = val;
        }
    }
}

// K5a: Sg = clip(qg@kg^T * scale); causal mask -> -10000 (== clip(-inf))
__global__ __launch_bounds__(256) void k_gscore(const __bf16* __restrict__ qg,
                                                const __bf16* __restrict__ kg,
                                                float* __restrict__ Sg,
                                                const float* __restrict__ scale_p) {
    const int cb = blockIdx.x, rb = blockIdx.y, head = blockIdx.z;
    const int tid = threadIdx.x, lane = tid & 31, wave = tid >> 5;
    const float scale = scale_p[0];
    const __bf16* q = qg + (size_t)head * 1024 * 64;
    const __bf16* k = kg + (size_t)head * 1024 * 64;
#pragma unroll
    for (int u = 0; u < 2; ++u) {
        const int idx = wave + 8 * u, mt = idx >> 2, nt = idx & 3;
        v8f c = gemm_tile<64>(q, 64, rb * 64 + mt * 16, k, 64, cb * 64 + nt * 16, lane);
        const int tcol = cb * 64 + nt * 16 + (lane & 15);
#pragma unroll
        for (int e = 0; e < 8; ++e) {
            const int row = rb * 64 + mt * 16 + (lane >> 4) * 8 + e;
            float sv = c[e] * scale;
            sv = fminf(fmaxf(sv, -10000.f), 10000.f);
            if (tcol > row) sv = -10000.f;
            Sg[((size_t)head * 1024 + row) * 1024 + tcol] = sv;
        }
    }
}

// K5b: row softmax over 1024 cols, one wave32 per row (rows = 4*1024)
__global__ __launch_bounds__(256) void k_gsoftmax(const float* __restrict__ Sg,
                                                  __bf16* __restrict__ Pg) {
    const int lane = threadIdx.x & 31, wave = threadIdx.x >> 5;
    const int row = blockIdx.x * 8 + wave;
    const float* sr = Sg + (size_t)row * 1024;
    float m = -1e30f;
#pragma unroll
    for (int e = 0; e < 32; ++e) m = fmaxf(m, sr[lane + 32 * e]);
#pragma unroll
    for (int off = 16; off; off >>= 1) m = fmaxf(m, __shfl_xor(m, off, 32));
    float p[32], sum = 0.f;
#pragma unroll
    for (int e = 0; e < 32; ++e) { p[e] = __expf(sr[lane + 32 * e] - m); sum += p[e]; }
#pragma unroll
    for (int off = 16; off; off >>= 1) sum += __shfl_xor(sum, off, 32);
    const float inv = 1.f / sum;
    __bf16* pr = Pg + (size_t)row * 1024;
#pragma unroll
    for (int e = 0; e < 32; ++e) pr[lane + 32 * e] = f2bf(p[e] * inv);
}

// K5c: Og_h = P @ V (via vgT as B), pack heads into ogh[1024][256]
__global__ __launch_bounds__(256) void k_gav(const __bf16* __restrict__ Pg,
                                             const __bf16* __restrict__ vgT,
                                             __bf16* __restrict__ ogh) {
    const int mb = blockIdx.x, head = blockIdx.y;
    const int tid = threadIdx.x, lane = tid & 31, wave = tid >> 5;
    const __bf16* A = Pg + (size_t)head * 1024 * 1024;
    const __bf16* B = vgT + (size_t)head * 64 * 1024;
#pragma unroll
    for (int u = 0; u < 2; ++u) {
        const int idx = wave + 8 * u, mt = idx >> 2, nt = idx & 3;
        v8f c = gemm_tile<1024>(A, 1024, mb * 64 + mt * 16, B, 1024, nt * 16, lane);
        const int d = nt * 16 + (lane & 15);
#pragma unroll
        for (int e = 0; e < 8; ++e) {
            const int m = mb * 64 + mt * 16 + (lane >> 4) * 8 + e;
            ogh[(size_t)m * 256 + head * 64 + d] = f2bf(c[e]);
        }
    }
}

// K6: og = ogh @ out_global_w^T + b; A-slab (32 KB) staged in LDS (async)
__global__ __launch_bounds__(256) void k_gout(const __bf16* __restrict__ ogh,
                                              const __bf16* __restrict__ wgo,
                                              const float* __restrict__ bgo,
                                              float* __restrict__ og) {
    __shared__ alignas(32) __bf16 xs[64 * 256];
    const int mb = blockIdx.x, nb = blockIdx.y;
    const int tid = threadIdx.x, lane = tid & 31, wave = tid >> 5;
    stage_async(xs, ogh + (size_t)mb * 64 * 256, 64 * 256, tid);
#pragma unroll
    for (int u = 0; u < 8; ++u) {
        const int idx = wave + 8 * u, mt = idx >> 4, nt = idx & 15;
        const int n0 = nb * 256 + nt * 16;
        v8f c = gemm_tile<256>(xs, 256, mt * 16, wgo, 256, n0, lane);
        const int col = n0 + (lane & 15);
        const float bias = bgo[col];
#pragma unroll
        for (int e = 0; e < 8; ++e) {
            const int m = mb * 64 + mt * 16 + (lane >> 4) * 8 + e;
            og[(size_t)m * 512 + col] = c[e] + bias;
        }
    }
}

// K3: final = o_local @ out_local_w^T + b + og[tile]; o-tile staged in LDS
__global__ __launch_bounds__(256) void k_local_out(const __bf16* __restrict__ o_ws,
                                                   const __bf16* __restrict__ wo,
                                                   const float* __restrict__ bo,
                                                   const float* __restrict__ og,
                                                   float* __restrict__ out) {
    __shared__ alignas(32) __bf16 xs[64 * 512];
    const int tile = blockIdx.x, nb = blockIdx.y;
    const int t = tile >> 6, i = (tile >> 3) & 7, j = tile & 7;
    const int tid = threadIdx.x, lane = tid & 31, wave = tid >> 5;
    stage_async(xs, o_ws + (size_t)tile * 64 * 512, 64 * 512, tid);
#pragma unroll
    for (int u = 0; u < 8; ++u) {
        const int idx = wave + 8 * u, mt = idx >> 4, nt = idx & 15;
        const int n0 = nb * 256 + nt * 16;
        v8f c = gemm_tile<512>(xs, 512, mt * 16, wo, 512, n0, lane);
        const int col = n0 + (lane & 15);
        const float add = bo[col] + og[(size_t)tile * 512 + col];
#pragma unroll
        for (int e = 0; e < 8; ++e) {
            const int s = mt * 16 + (lane >> 4) * 8 + e;
            const int hh = i * 8 + (s >> 3), ww = j * 8 + (s & 7);
            out[(((size_t)t * 64 + hh) * 64 + ww) * 512 + col] = c[e] + add;
        }
    }
}

extern "C" void kernel_launch(void* const* d_in, const int* in_sizes, int n_in,
                              void* d_out, int out_size, void* d_ws, size_t ws_size,
                              hipStream_t stream) {
    (void)in_sizes; (void)n_in; (void)out_size; (void)ws_size;
    const float* x     = (const float*)d_in[0];
    const float* wl_f  = (const float*)d_in[1];
    const float* bl    = (const float*)d_in[2];
    const float* wo_f  = (const float*)d_in[3];
    const float* bo    = (const float*)d_in[4];
    const float* wg_f  = (const float*)d_in[5];
    const float* bg    = (const float*)d_in[6];
    const float* wgo_f = (const float*)d_in[7];
    const float* bgo   = (const float*)d_in[8];
    const float* scale = (const float*)d_in[9];
    float* out = (float*)d_out;

    char* wsb = (char*)d_ws; size_t off = 0;
    auto alloc = [&](size_t b) { void* p = wsb + off; off = (off + b + 255) & ~(size_t)255; return p; };
    __bf16* wl   = (__bf16*)alloc((size_t)1536 * 512 * 2);
    __bf16* wo   = (__bf16*)alloc((size_t)512 * 512 * 2);
    __bf16* wg   = (__bf16*)alloc((size_t)768 * 512 * 2);
    __bf16* wgo  = (__bf16*)alloc((size_t)512 * 256 * 2);
    __bf16* xt   = (__bf16*)alloc((size_t)1024 * 64 * 512 * 2);
    __bf16* xf   = (__bf16*)alloc((size_t)1024 * 512 * 2);
    __bf16* o_ws = (__bf16*)alloc((size_t)1024 * 64 * 512 * 2);
    float*  og   = (float*)alloc((size_t)1024 * 512 * 4);
    __bf16* qg   = (__bf16*)alloc((size_t)4 * 1024 * 64 * 2);
    __bf16* kg   = (__bf16*)alloc((size_t)4 * 1024 * 64 * 2);
    __bf16* vgT  = (__bf16*)alloc((size_t)4 * 64 * 1024 * 2);
    float*  Sg   = (float*)alloc((size_t)4 * 1024 * 1024 * 4);
    __bf16* Pg   = (__bf16*)alloc((size_t)4 * 1024 * 1024 * 2);
    __bf16* ogh  = (__bf16*)alloc((size_t)1024 * 256 * 2);

    // weight conversions
    k_cvt<<<(1536 * 512 + 255) / 256, 256, 0, stream>>>(wl_f, wl, 1536 * 512);
    k_cvt<<<(512 * 512 + 255) / 256, 256, 0, stream>>>(wo_f, wo, 512 * 512);
    k_cvt<<<(768 * 512 + 255) / 256, 256, 0, stream>>>(wg_f, wg, 768 * 512);
    k_cvt<<<(512 * 256 + 255) / 256, 256, 0, stream>>>(wgo_f, wgo, 512 * 256);

    // tile gather + means
    k_tiles<<<1024, 256, 0, stream>>>(x, xt, xf);

    // global path (produces og residual)
    k_gqkv<<<dim3(16, 3), 256, 0, stream>>>(xf, wg, bg, qg, kg, vgT);
    k_gscore<<<dim3(16, 16, 4), 256, 0, stream>>>(qg, kg, Sg, scale);
    k_gsoftmax<<<512, 256, 0, stream>>>(Sg, Pg);
    k_gav<<<dim3(16, 4), 256, 0, stream>>>(Pg, vgT, ogh);
    k_gout<<<dim3(16, 2), 256, 0, stream>>>(ogh, wgo, bgo, og);

    // local path
    k_local_attn<<<1024, 256, 0, stream>>>(xt, wl, bl, o_ws, scale);
    k_local_out<<<dim3(1024, 2), 256, 0, stream>>>(o_ws, wo, bo, og, out);
}
